// AggMaskStarHead_52888227283366
// MI455X (gfx1250) — matrix operand order repaired and checked by reading, not compiled, not run on hardware
//
#include <hip/hip_runtime.h>
#include <hip/hip_bf16.h>
#include <math.h>

// ---------------------------------------------------------------------------
// MI455X (gfx1250) implementation. wave32, WMMA f32_16x16x32_f16 for every
// GEMM. Memory-bound workload (~135 MB HBM, ~5.8 GF): resize/matmul commuted
// for ctx, feat staged as f16, and the dominant dynamic-MLP stage stages a
// 10 KB feat slab into LDS per workgroup via async-to-LDS (ASYNCcnt) so the
// 288x feat reuse is served from LDS instead of L2.
// ---------------------------------------------------------------------------

typedef _Float16 v16h __attribute__((ext_vector_type(16)));
typedef float    v8f  __attribute__((ext_vector_type(8)));
typedef int      v4i  __attribute__((ext_vector_type(4)));

#define C_IN    256
#define C_CTX   16
#define NN      4
#define C_MID   8
#define S_MASK  24
#define S_CLS   40
#define NUM_CLS 80
#define CIN0    (NN + C_CTX)       /* 20 */
#define LW_NUM  249
#define NPOS    (S_MASK * S_MASK)  /* 576 */
#define HW0     (128 * 128)        /* 16384 */
#define KCONV   (C_IN * 9)         /* 2304 */

#if defined(__gfx1250__) && __has_builtin(__builtin_amdgcn_global_load_async_to_lds_b128)
#define HAVE_ASYNC_LDS 1
// builtin signature (from hipcc diagnostic): (v4i AS1*, v4i AS3*, imm, imm)
typedef __attribute__((address_space(1))) v4i as1_v4i;
typedef __attribute__((address_space(3))) v4i as3_v4i;
#endif

// A-matrix (16x32 f16) per-lane element -> K index (contiguous, matches B):
// lanes 0-15: K0..15 ; lanes 16-31: K16..31
__device__ __forceinline__ int a_k_of(int lane, int e) {
  return e + ((lane & 16) ? 16 : 0);
}

__device__ __forceinline__ v8f wmma_f16(v16h a, v16h b, v8f c) {
  // (neg_a, A, neg_b, B, c_mod, C, reuse_a, reuse_b)
  return __builtin_amdgcn_wmma_f32_16x16x32_f16(false, a, false, b, (short)0, c,
                                                false, false);
}

// jax.image.resize(method='bilinear', antialias=True) compatible sampler:
// triangle kernel, width scaled by 1/scale on downsample, edge-renormalized.
__device__ __forceinline__ float jax_resample(const float* __restrict__ src,
                                              int SH, int SW, int dy, int dx,
                                              int DH, int DW) {
  float scy = (float)DH / (float)SH;
  float scx = (float)DW / (float)SW;
  float cy = (dy + 0.5f) / scy - 0.5f;
  float cx = (dx + 0.5f) / scx - 0.5f;
  float ky = scy < 1.f ? scy : 1.f;
  float kx = scx < 1.f ? scx : 1.f;
  float ry = 1.f / ky, rx = 1.f / kx;
  int iy0 = max(0, (int)ceilf(cy - ry));
  int iy1 = min(SH - 1, (int)floorf(cy + ry));
  int ix0 = max(0, (int)ceilf(cx - rx));
  int ix1 = min(SW - 1, (int)floorf(cx + rx));
  float acc = 0.f, wsum = 0.f;
  for (int iy = iy0; iy <= iy1; ++iy) {
    float wy = 1.f - fabsf((iy - cy) * ky);
    if (wy <= 0.f) continue;
    for (int ix = ix0; ix <= ix1; ++ix) {
      float wx = 1.f - fabsf((ix - cx) * kx);
      if (wx <= 0.f) continue;
      float w = wy * wx;
      acc += w * src[iy * SW + ix];
      wsum += w;
    }
  }
  return wsum > 0.f ? acc / wsum : 0.f;
}

// ---------------------------------------------------------------------------
// Stage A: y_s = W_s(16x256) @ x_s(256xHW).  One 16-pixel tile per wave,
// K=256 in 8 WMMA chunks. (resize commutes with channel matmul -> done later.)
// ---------------------------------------------------------------------------
__global__ void level_gemm_kernel(const float* __restrict__ x,
                                  const float* __restrict__ wctx, int koff0,
                                  float* __restrict__ y, int HW) {
  int lane = threadIdx.x & 31;
  int wid = blockIdx.x * (blockDim.x >> 5) + (threadIdx.x >> 5);
  int ntiles = HW >> 4;
  if (wid >= ntiles) return;               // wave-uniform: EXEC full for WMMA
  int n = (wid << 4) + (lane & 15);
  int kb = (lane & 16) ? 16 : 0;
  v8f acc = {};
  for (int kc = 0; kc < C_IN; kc += 32) {
    v16h a, b;
#pragma unroll
    for (int e = 0; e < 16; ++e) {
      int ka = kc + a_k_of(lane, e);
      a[e] = (_Float16)wctx[(lane & 15) * (C_IN * 5) + koff0 + ka];
      b[e] = (_Float16)x[(kc + kb + e) * HW + n];
    }
    acc = wmma_f16(a, b, acc);
  }
  int mbase = (lane & 16) ? 8 : 0;
#pragma unroll
  for (int r = 0; r < 8; ++r) y[(mbase + r) * HW + n] = acc[r];
}

// ---------------------------------------------------------------------------
// Generic resize to f16 staging buffer (kin: x2->24x24, cin: x0->40x40).
// ---------------------------------------------------------------------------
__global__ void resample_f16_kernel(const float* __restrict__ src, int C,
                                    int SH, int SW, _Float16* __restrict__ dst,
                                    int DH, int DW) {
  int idx = blockIdx.x * blockDim.x + threadIdx.x;
  int total = C * DH * DW;
  if (idx >= total) return;
  int c = idx / (DH * DW);
  int r = idx % (DH * DW);
  dst[idx] =
      (_Float16)jax_resample(src + c * SH * SW, SH, SW, r / DW, r % DW, DH, DW);
}

// ---------------------------------------------------------------------------
// ctx = relu(y0 + sum_s up(y_s) + b_ctx); feat f16 = [mask_feat(4), ctx(16)].
// ---------------------------------------------------------------------------
__global__ void ctx_feat_kernel(const float* __restrict__ y0,
                                const float* __restrict__ y1,
                                const float* __restrict__ y2,
                                const float* __restrict__ y3,
                                const float* __restrict__ y4,
                                const float* __restrict__ mask_feat,
                                const float* __restrict__ b_ctx,
                                _Float16* __restrict__ feat) {
  int p = blockIdx.x * blockDim.x + threadIdx.x;
  if (p >= HW0) return;
  int py = p >> 7, px = p & 127;
#pragma unroll
  for (int c = 0; c < NN; ++c) feat[c * HW0 + p] = (_Float16)mask_feat[c * HW0 + p];
  const float* ys[4] = {y1, y2, y3, y4};
  const int dim[4] = {64, 32, 16, 8};
  for (int m = 0; m < C_CTX; ++m) {
    float v = y0[m * HW0 + p] + b_ctx[m];
    for (int s = 0; s < 4; ++s) {
      int D = dim[s];
      v += jax_resample(ys[s] + m * D * D, D, D, py, px, 128, 128);
    }
    v = v > 0.f ? v : 0.f;
    feat[(NN + m) * HW0 + p] = (_Float16)v;
  }
}

// ---------------------------------------------------------------------------
// 3x3 SAME conv as im2col GEMM (M x 2304 x H*W) via WMMA. B built on the fly
// from f16 staged src with zero padding. Used for kp (M=249) and cate (M=80).
// ---------------------------------------------------------------------------
__global__ void conv3x3_gemm_kernel(const _Float16* __restrict__ src, int H,
                                    int W, const float* __restrict__ wt,
                                    const float* __restrict__ bias, int Mreal,
                                    float* __restrict__ out) {
  int lane = threadIdx.x & 31;
  int wid = blockIdx.x * (blockDim.x >> 5) + (threadIdx.x >> 5);
  int N = H * W;
  int ntiles = N >> 4;                     // N = 576 or 1600, multiple of 16
  int mtiles = (Mreal + 15) >> 4;
  if (wid >= mtiles * ntiles) return;      // wave-uniform
  int mt = wid / ntiles, nt = wid % ntiles;
  int n = (nt << 4) + (lane & 15);
  int pyb = n / W, pxb = n % W;
  int mg = (mt << 4) + (lane & 15);        // A-row this lane feeds
  int kb = (lane & 16) ? 16 : 0;
  v8f acc = {};
  for (int kc = 0; kc < KCONV; kc += 32) {
    v16h a, b;
#pragma unroll
    for (int e = 0; e < 16; ++e) {
      int ka = kc + a_k_of(lane, e);       // k = c*9 + ky*3 + kx (OIHW flat)
      a[e] = (mg < Mreal) ? (_Float16)wt[mg * KCONV + ka] : (_Float16)0.f;
      int kk = kc + kb + e;
      int c = kk / 9, t = kk % 9;
      int sy = pyb + t / 3 - 1, sx = pxb + t % 3 - 1;
      b[e] = (sy >= 0 && sy < H && sx >= 0 && sx < W)
                 ? src[(c * H + sy) * W + sx]
                 : (_Float16)0.f;
    }
    acc = wmma_f16(a, b, acc);
  }
  int mbase = (mt << 4) + ((lane & 16) ? 8 : 0);
#pragma unroll
  for (int r = 0; r < 8; ++r) {
    int m = mbase + r;
    if (m < Mreal) out[m * N + n] = acc[r] + bias[m];
  }
}

// ---------------------------------------------------------------------------
// Dynamic per-position MLP. Grid = 64 pixel-slabs x 4 pair-groups.
// Per block: async-copy a 20x256 f16 feat slab (10 KB) into LDS
// (GLOBAL_LOAD_ASYNC_TO_LDS_B128, ASYNCcnt), barrier, then 8 waves x 9
// position-pairs x 16 pixel-tiles of WMMA with B read from LDS (DS ops).
// 2 positions per WMMA tile: C layout leaves all 8 layer-1 channels of one
// position in one lane's accumulators -> layers 2/3 are per-lane FMAs with
// half-wave-uniform (broadcast) weight loads.
// ---------------------------------------------------------------------------
#define PIXG 256
#define PAIR_GROUPS 4
#define PAIRS_PER_WAVE 9   /* 288 pairs / (4 groups * 8 waves) */

__global__ void dyn_mask_kernel(const _Float16* __restrict__ feat,
                                const float* __restrict__ kp,
                                float* __restrict__ out) {
  __shared__ __align__(16) _Float16 sfeat[CIN0 * PIXG];  // 10 KB LDS slab
  int tid = threadIdx.x;
  int lane = tid & 31;
  int wv = tid >> 5;                       // 0..7
  int pairgrp = blockIdx.x & (PAIR_GROUPS - 1);
  int pixbase = (blockIdx.x >> 2) * PIXG;  // 64 slabs

  // ---- cooperative slab copy: global feat[k][pixbase..+256) -> LDS --------
  const int NB128 = CIN0 * PIXG / 8;       // 640 16-byte transfers
  for (int i = tid; i < NB128; i += 256) {
    int k = i >> 5;                        // / (PIXG/8)
    int j = (i & 31) << 3;                 // half-element offset in row
#ifdef HAVE_ASYNC_LDS
    __builtin_amdgcn_global_load_async_to_lds_b128(
        (as1_v4i*)(feat + (size_t)k * HW0 + pixbase + j),
        (as3_v4i*)(&sfeat[k * PIXG + j]), 0, 0);
#else
#pragma unroll
    for (int q = 0; q < 8; ++q)
      sfeat[k * PIXG + j + q] = feat[(size_t)k * HW0 + pixbase + j + q];
#endif
  }
#ifdef HAVE_ASYNC_LDS
#if __has_builtin(__builtin_amdgcn_s_wait_asynccnt)
  __builtin_amdgcn_s_wait_asynccnt(0);     // wait own wave's async loads
#else
  asm volatile("s_wait_asynccnt 0x0" ::: "memory");
#endif
#endif
  __syncthreads();                         // all waves' slab data visible

  int kb = (lane & 16) ? 16 : 0;
  for (int pp = 0; pp < PAIRS_PER_WAVE; ++pp) {
    int pair = pairgrp * (PAIRS_PER_WAVE * 8) + wv * PAIRS_PER_WAVE + pp;
    int pos = pair * 2 + ((lane & 16) ? 1 : 0);
    // hint next pair's layer-2 params toward cache (global_prefetch_b8)
    __builtin_prefetch(kp + 168 * NPOS + ((pair + 1) & 287) * 2, 0, 0);
    // per-lane layer 2/3 params (kp layout: [param][position])
    float b0l[8], w1l[64], b1l[8], w2l[8];
#pragma unroll
    for (int i = 0; i < 8; ++i) b0l[i] = kp[(C_MID * CIN0 + i) * NPOS + pos];
#pragma unroll
    for (int i = 0; i < 64; ++i) w1l[i] = kp[(168 + i) * NPOS + pos];
#pragma unroll
    for (int i = 0; i < 8; ++i) b1l[i] = kp[(232 + i) * NPOS + pos];
#pragma unroll
    for (int i = 0; i < 8; ++i) w2l[i] = kp[(240 + i) * NPOS + pos];
    float b2l = kp[248 * NPOS + pos];
    // A fragment: rows 0..7 = pos_a W0, rows 8..15 = pos_b W0, K 20 -> 32 pad
    int m = lane & 15;
    int pos_m = pair * 2 + (m >> 3);
    int om = m & 7;
    v16h a;
#pragma unroll
    for (int e = 0; e < 16; ++e) {
      int k = a_k_of(lane, e);
      a[e] = (k < CIN0) ? (_Float16)kp[(om * CIN0 + k) * NPOS + pos_m]
                        : (_Float16)0.f;
    }
    for (int t = 0; t < PIXG / 16; ++t) {
      int pix_local = t * 16 + (lane & 15);
      v16h b;
#pragma unroll
      for (int e = 0; e < 16; ++e) {       // B from LDS (ds_load)
        int k = kb + e;
        b[e] = (k < CIN0) ? sfeat[k * PIXG + pix_local] : (_Float16)0.f;
      }
      v8f c = {};
      c = wmma_f16(a, b, c);
      float h[8];
#pragma unroll
      for (int i = 0; i < 8; ++i) {
        float v = c[i] + b0l[i];
        h[i] = v > 0.f ? v : 0.f;
      }
      float mv = b2l;
#pragma unroll
      for (int j = 0; j < 8; ++j) {
        float v = b1l[j];
#pragma unroll
        for (int i = 0; i < 8; ++i) v += w1l[j * 8 + i] * h[i];
        v = v > 0.f ? v : 0.f;
        mv += w2l[j] * v;
      }
      out[(size_t)pos * HW0 + pixbase + pix_local] =
          1.f / (1.f + __expf(-mv));
    }
  }
}

// ---------------------------------------------------------------------------
// sigmoid + 2x2 points-NMS (compare pre-sigmoid logits: monotonic).
// hmax[i,j] = max(heat[i-1..i, j-1..j]); keep iff heat == hmax.
// ---------------------------------------------------------------------------
__global__ void cate_nms_kernel(const float* __restrict__ cate,
                                float* __restrict__ out) {
  int idx = blockIdx.x * blockDim.x + threadIdx.x;
  if (idx >= NUM_CLS * S_CLS * S_CLS) return;
  int r = idx % (S_CLS * S_CLS);
  int i = r / S_CLS, j = r % S_CLS;
  const float* pc = cate + (idx - r);
  float v = pc[r];
  float mx = -3.4e38f;
  if (i > 0) mx = fmaxf(mx, pc[r - S_CLS]);
  if (j > 0) mx = fmaxf(mx, pc[r - 1]);
  if (i > 0 && j > 0) mx = fmaxf(mx, pc[r - S_CLS - 1]);
  float s = 1.f / (1.f + __expf(-v));
  out[idx] = (v >= mx) ? s : 0.f;
}

// ---------------------------------------------------------------------------
extern "C" void kernel_launch(void* const* d_in, const int* in_sizes, int n_in,
                              void* d_out, int out_size, void* d_ws,
                              size_t ws_size, hipStream_t stream) {
  const float* x0 = (const float*)d_in[0];
  const float* x1 = (const float*)d_in[1];
  const float* x2 = (const float*)d_in[2];
  const float* x3 = (const float*)d_in[3];
  const float* x4 = (const float*)d_in[4];
  const float* mask_feat = (const float*)d_in[5];
  const float* w_ctx = (const float*)d_in[6];
  const float* b_ctx = (const float*)d_in[7];
  const float* w_lw = (const float*)d_in[8];
  const float* b_lw = (const float*)d_in[9];
  const float* w_cate = (const float*)d_in[10];
  const float* b_cate = (const float*)d_in[11];
  (void)in_sizes; (void)n_in; (void)out_size; (void)ws_size;

  char* ws = (char*)d_ws;
  size_t off = 0;
  auto alloc = [&](size_t bytes) {
    size_t r = off;
    off = (off + bytes + 255) & ~(size_t)255;
    return r;
  };
  float* y0 = (float*)(ws + alloc(16 * 16384 * 4));
  float* y1 = (float*)(ws + alloc(16 * 4096 * 4));
  float* y2 = (float*)(ws + alloc(16 * 1024 * 4));
  float* y3 = (float*)(ws + alloc(16 * 256 * 4));
  float* y4 = (float*)(ws + alloc(16 * 64 * 4));
  _Float16* feat = (_Float16*)(ws + alloc(20 * 16384 * 2));
  _Float16* kin = (_Float16*)(ws + alloc(256 * NPOS * 2));
  float* kp = (float*)(ws + alloc((size_t)LW_NUM * NPOS * 4));
  _Float16* cin = (_Float16*)(ws + alloc(256 * 1600 * 2));
  float* cate = (float*)(ws + alloc(80 * 1600 * 4));

  // Stage A: per-level channel GEMMs at native resolution (resize commuted)
  level_gemm_kernel<<<128, 256, 0, stream>>>(x0, w_ctx, 0, y0, 16384);
  level_gemm_kernel<<<32, 256, 0, stream>>>(x1, w_ctx, 256, y1, 4096);
  level_gemm_kernel<<<8, 256, 0, stream>>>(x2, w_ctx, 512, y2, 1024);
  level_gemm_kernel<<<2, 256, 0, stream>>>(x3, w_ctx, 768, y3, 256);
  level_gemm_kernel<<<1, 256, 0, stream>>>(x4, w_ctx, 1024, y4, 64);

  // f16 staging of conv inputs
  resample_f16_kernel<<<(256 * NPOS + 255) / 256, 256, 0, stream>>>(
      x2, 256, 32, 32, kin, 24, 24);
  resample_f16_kernel<<<(256 * 1600 + 255) / 256, 256, 0, stream>>>(
      x0, 256, 128, 128, cin, 40, 40);

  // ctx upsample-combine + feat(f16) build
  ctx_feat_kernel<<<HW0 / 256, 256, 0, stream>>>(y0, y1, y2, y3, y4, mask_feat,
                                                 b_ctx, feat);

  // kernel-predictor conv (249ch) and category conv (80ch) as WMMA GEMMs
  conv3x3_gemm_kernel<<<(16 * 36 + 7) / 8, 256, 0, stream>>>(kin, 24, 24, w_lw,
                                                             b_lw, LW_NUM, kp);
  conv3x3_gemm_kernel<<<(5 * 100 + 7) / 8, 256, 0, stream>>>(
      cin, 40, 40, w_cate, b_cate, NUM_CLS, cate);

  // dynamic MLP: 64 pixel slabs x 4 pair groups (dominant stage)
  dyn_mask_kernel<<<64 * PAIR_GROUPS, 256, 0, stream>>>(feat, kp,
                                                        (float*)d_out);

  // cate sigmoid + points-NMS appended after seg_masks in d_out
  cate_nms_kernel<<<(NUM_CLS * S_CLS * S_CLS + 255) / 256, 256, 0, stream>>>(
      cate, (float*)d_out + (size_t)NPOS * HW0);
}
